// UpsampleLayer_77884936945771
// MI455X (gfx1250) — matrix operand order repaired and checked by reading, not compiled, not run on hardware
//
#include <hip/hip_runtime.h>
#include <hip/hip_bf16.h>

// MI455X / gfx1250, wave32. Upsample2x(nearest)+Conv3x3+bias as parity GEMMs
// (M=128, K=512 per row-parity pair) on v_wmma_f32_16x16x32_bf16, with TDM
// (tensor_load_to_lds) staging of a pre-padded channel-last bf16 tensor.

typedef __attribute__((ext_vector_type(16))) __bf16       v16bf;
typedef __attribute__((ext_vector_type(8)))  __bf16       v8bf;
typedef __attribute__((ext_vector_type(8)))  float        v8f;
typedef __attribute__((ext_vector_type(4)))  unsigned int u32x4;
typedef __attribute__((ext_vector_type(8)))  int          i32x8;
typedef __attribute__((ext_vector_type(4)))  int          i32x4;

#define CIN    128
#define COUT   128
#define HIN    64
#define WIN    64
#define NBATCH 8
#define XPAD   66                 // padded spatial extent (1-px zero border)
#define LDS_CI 136                // ci stride in LDS: 256B data + 16B pad
#define XT_PLANE ((size_t)XPAD * XPAD * CIN)            // per-batch elements
#define XT_BYTES ((size_t)NBATCH * XT_PLANE * 2)        // 8,921,088
#define WA_OFF   XT_BYTES                               // WA after xT in ws

// ---------------------------------------------------------------------------
// Kernel A: x [B][C][64][64] f32  ->  xT [B][yp 66][xp 66][ci 128] bf16,
// zero border. LDS transpose per (b,yp) row for coalesced in and out.
// ---------------------------------------------------------------------------
__global__ __launch_bounds__(256) void xpad_kernel(
    const float* __restrict__ x, __bf16* __restrict__ xT) {
  __shared__ __align__(16) __bf16 tile[XPAD][CIN];      // 16,896 B
  int bid = blockIdx.x;                                  // 0..527
  int yp  = bid % XPAD;
  int b   = bid / XPAD;
  int t   = threadIdx.x;

  if (yp >= 1 && yp <= HIN) {
    int ci = t >> 1, half = t & 1;
    const float* src = x + (((size_t)b * CIN + ci) * HIN + (yp - 1)) * WIN + half * 32;
#pragma unroll
    for (int j = 0; j < 32; ++j)
      tile[1 + half * 32 + j][ci] = (__bf16)src[j];
    if (t < CIN) { tile[0][t] = (__bf16)0.f; tile[XPAD - 1][t] = (__bf16)0.f; }
  } else {
    v8bf z = {};
    for (int i = t; i < XPAD * CIN / 8; i += 256)
      ((v8bf*)&tile[0][0])[i] = z;
  }
  __syncthreads();

  __bf16* dst = xT + ((size_t)b * XPAD + yp) * XPAD * CIN;
  // 16,896 B = 1056 x 16B, fully coalesced
  for (int i = t; i < XPAD * CIN / 8; i += 256)
    ((uint4*)dst)[i] = ((const uint4*)&tile[0][0])[i];
}

// ---------------------------------------------------------------------------
// Kernel B: fold conv3x3 into 4 parity 2x2 kernels, bf16, pre-swizzled into
// the WMMA A-fragment (16x32, 16-bit) per-lane layout (ISA 7.12.2).
// WA[p][kk][mtile][lane] : v16bf   (4*16*8*32 * 32B = 512 KB)
//   K (0..511) = tap*128 + ci,  tap = ay*2 + bx.
// ---------------------------------------------------------------------------
__global__ __launch_bounds__(256) void upconv_wxform(
    const float* __restrict__ W, __bf16* __restrict__ wa) {
  int t    = blockIdx.x * blockDim.x + threadIdx.x;     // 0..16383
  int lane = t & 31;
  int mt   = (t >> 5) & 7;
  int kk   = (t >> 8) & 15;
  int p    = (t >> 12) & 3;
  int py = p >> 1, px = p & 1;
  int M  = mt * 16 + (lane & 15);
  int lh = (lane >> 4) & 1;

  v16bf frag;
#pragma unroll
  for (int e = 0; e < 16; ++e) {
    int Kl  = (e & 7) + ((e >= 8) ? 16 : 0) + lh * 8;
    int Kg  = kk * 32 + Kl;
    int tap = Kg >> 7;
    int ci  = Kg & 127;
    int ay = tap >> 1, bx = tap & 1;
    int ky0, ky1, kx0, kx1;
    if (py == 0) { if (ay == 0) { ky0 = 0; ky1 = -1; } else { ky0 = 1; ky1 = 2; } }
    else         { if (ay == 0) { ky0 = 0; ky1 =  1; } else { ky0 = 2; ky1 = -1; } }
    if (px == 0) { if (bx == 0) { kx0 = 0; kx1 = -1; } else { kx0 = 1; kx1 = 2; } }
    else         { if (bx == 0) { kx0 = 0; kx1 =  1; } else { kx0 = 2; kx1 = -1; } }
    const float* wb = W + ((size_t)M * CIN + ci) * 9;
    float v = 0.f;
#pragma unroll
    for (int yi = 0; yi < 2; ++yi) {
      int ky = (yi == 0) ? ky0 : ky1;
      if (ky < 0) continue;
#pragma unroll
      for (int xi = 0; xi < 2; ++xi) {
        int kx = (xi == 0) ? kx0 : kx1;
        if (kx < 0) continue;
        v += wb[ky * 3 + kx];
      }
    }
    frag[e] = (__bf16)v;
  }
  ((v16bf*)wa)[t] = frag;
}

// ---------------------------------------------------------------------------
// TDM: DMA `nrows` contiguous 256B ci-rows from xT into LDS, inserting 16B of
// padding after every 256B (-> LDS ci stride 272B). D# per ISA 8.3/8.4:
//   g0: count=1 | lds_addr | global_addr(57b) | type=2
//   g1: data_size=2B, pad_enable, pad_interval=64dw(code 5), pad_amount=4dw(code 3)
//       1-D tile: tensor_dim0 = tile_dim0 = nrows*128 elements
// ---------------------------------------------------------------------------
__device__ __forceinline__ void tdm_load_rows(unsigned lds_byte,
                                              const __bf16* gsrc, int nrows) {
  unsigned long long ga = (unsigned long long)(uintptr_t)gsrc;
  unsigned tile0 = (unsigned)nrows * 128u;              // elements (<=2176)
  u32x4 g0;
  g0[0] = 1u;                                           // count=1, normal mode
  g0[1] = lds_byte;                                     // lds_addr
  g0[2] = (unsigned)(ga & 0xFFFFFFFFu);                 // global_addr[31:0]
  g0[3] = (unsigned)((ga >> 32) & 0x01FFFFFFu) | 0x80000000u; // [56:32]|type=2
  i32x8 g1;
  g1[0] = (int)((1u << 16) | (1u << 20) | (5u << 22) | (3u << 25));
  g1[1] = (int)((tile0 & 0xFFFFu) << 16);               // tensor_dim0 lo16
  g1[2] = (int)((tile0 >> 16) | (1u << 16));            // tensor_dim0 hi16 | tensor_dim1=1
  g1[3] = (int)((tile0 & 0xFFFFu) << 16);               // tile_dim0
  g1[4] = 0;                                            // tile_dim1/2 unused
  g1[5] = (int)tile0;                                   // tensor_dim0_stride (unused, sane)
  g1[6] = 0;
  g1[7] = 0;
  i32x4 z4 = {0, 0, 0, 0};
  i32x8 z8 = {0, 0, 0, 0, 0, 0, 0, 0};
  // clang-23 / therock-10.0 form: 6 args (g0, g1, g2, g3, g4, cpol)
  __builtin_amdgcn_tensor_load_to_lds(g0, g1, z4, z4, z8, 0);
}

// Load a 16 x bf16 B fragment: 8 elems at p and 8 elems at p+16, from a 16B-aligned LDS address.
__device__ __forceinline__ v16bf ldfrag(const __bf16* p) {
  const v8bf* q = (const v8bf*)p;
  v8bf lo = q[0], hi = q[2];
  return __builtin_shufflevector(lo, hi, 0,1,2,3,4,5,6,7,8,9,10,11,12,13,14,15);
}

// ---------------------------------------------------------------------------
// Kernel C: main implicit GEMM. Block = 256 thr (8 waves).
// bid -> (py, b, iy). Block computes BOTH px parities for one output row pair:
// 128 C_out x 64 px-pairs. Wave w: C_out 16w..16w+15, 4 N-tiles x 2 parities.
// ---------------------------------------------------------------------------
__global__ __launch_bounds__(256) void upconv_main(
    const __bf16* __restrict__ xT, const __bf16* __restrict__ wa,
    const float* __restrict__ bias, float* __restrict__ out) {
  __shared__ __align__(16) __bf16 xs[2][XPAD][LDS_CI];  // 35,904 B

  int bid = blockIdx.x;                                  // 0..1023
  int iy  = bid & 63;
  int b   = (bid >> 6) & 7;
  int py  = (bid >> 9) & 1;
  int tid  = threadIdx.x;
  int w    = tid >> 5;
  int lane = tid & 31;
  int n    = lane & 15;
  int lh   = lane >> 4;

  // ---- TDM-stage slab: 132 ci-rows (2 padded y-rows x 66 xp), split 8 ways ----
  {
    int rs = w * 16 + (w < 4 ? w : 4);                   // 17,17,17,17,16,16,16,16
    int nr = (w < 4) ? 17 : 16;
    const __bf16* gsrc = xT + (size_t)b * XT_PLANE
                            + (size_t)(iy + py) * XPAD * CIN
                            + (size_t)rs * CIN;
    unsigned ldsb = (unsigned)(uintptr_t)(&xs[0][0][0]) + (unsigned)rs * (LDS_CI * 2);
    tdm_load_rows(ldsb, gsrc, nr);
    __builtin_amdgcn_s_wait_tensorcnt(0);
  }
  __syncthreads();

  v8f acc[2][4] = {};                                    // [px][ntile]
  const v16bf* WA = (const v16bf*)wa;
  int p0 = py * 2, p1 = py * 2 + 1;

  // Loop-invariant bases: A pointers (per-parity), LDS lane base. All per-kk
  // deltas become constant load-offset immediates under full unroll.
  const v16bf* ap0 = WA + (size_t)(p0 * 128 + w) * 32 + lane;   // + kk*256
  const v16bf* ap1 = WA + (size_t)(p1 * 128 + w) * 32 + lane;
  const __bf16* xb = &xs[0][0][0] + n * LDS_CI + lh * 8;        // lane-dependent base

#pragma unroll
  for (int kk = 0; kk < 16; ++kk) {
    const int tap = kk >> 2;
    const int ay  = tap >> 1;
    const int bx  = tap & 1;
    const int cb  = (kk & 3) * 32;                       // ci sub-chunk

    v16bf af0 = ap0[kk * 256];                           // 32B/lane, L2 hit
    v16bf af1 = ap1[kk * 256];

    // All 8 B fragments first (16 ds_load_b128 batched under one wait),
    // then 8 WMMAs. LDS offsets are compile-time immediates off xb.
    v16bf f00 = ldfrag(xb + ((ay * XPAD +  0 + bx)     * LDS_CI + cb));  // px0 nt0
    v16bf f01 = ldfrag(xb + ((ay * XPAD + 16 + bx)     * LDS_CI + cb));  // px0 nt1
    v16bf f02 = ldfrag(xb + ((ay * XPAD + 32 + bx)     * LDS_CI + cb));  // px0 nt2
    v16bf f03 = ldfrag(xb + ((ay * XPAD + 48 + bx)     * LDS_CI + cb));  // px0 nt3
    v16bf f10 = ldfrag(xb + ((ay * XPAD +  1 + bx)     * LDS_CI + cb));  // px1 nt0
    v16bf f11 = ldfrag(xb + ((ay * XPAD + 17 + bx)     * LDS_CI + cb));  // px1 nt1
    v16bf f12 = ldfrag(xb + ((ay * XPAD + 33 + bx)     * LDS_CI + cb));  // px1 nt2
    v16bf f13 = ldfrag(xb + ((ay * XPAD + 49 + bx)     * LDS_CI + cb));  // px1 nt3

    acc[0][0] = __builtin_amdgcn_wmma_f32_16x16x32_bf16(false, af0, false, f00,
                                                        (short)0, acc[0][0], false, false);
    acc[0][1] = __builtin_amdgcn_wmma_f32_16x16x32_bf16(false, af0, false, f01,
                                                        (short)0, acc[0][1], false, false);
    acc[0][2] = __builtin_amdgcn_wmma_f32_16x16x32_bf16(false, af0, false, f02,
                                                        (short)0, acc[0][2], false, false);
    acc[0][3] = __builtin_amdgcn_wmma_f32_16x16x32_bf16(false, af0, false, f03,
                                                        (short)0, acc[0][3], false, false);
    acc[1][0] = __builtin_amdgcn_wmma_f32_16x16x32_bf16(false, af1, false, f10,
                                                        (short)0, acc[1][0], false, false);
    acc[1][1] = __builtin_amdgcn_wmma_f32_16x16x32_bf16(false, af1, false, f11,
                                                        (short)0, acc[1][1], false, false);
    acc[1][2] = __builtin_amdgcn_wmma_f32_16x16x32_bf16(false, af1, false, f12,
                                                        (short)0, acc[1][2], false, false);
    acc[1][3] = __builtin_amdgcn_wmma_f32_16x16x32_bf16(false, af1, false, f13,
                                                        (short)0, acc[1][3], false, false);
  }

  // ---- epilogue: bias + (even,odd) pixel-pair f32x2 stores, fully coalesced ----
  int co_base = w * 16 + lh * 8;
  int oy = 2 * iy + py;
#pragma unroll
  for (int r = 0; r < 8; ++r) {
    int co = co_base + r;
    float bv = bias[co];
    size_t rowo = (((size_t)b * COUT + co) * 128 + oy) * 128;
#pragma unroll
    for (int nt = 0; nt < 4; ++nt) {
      int ox = 2 * (nt * 16 + n);
      float2 v;
      v.x = acc[0][nt][r] + bv;
      v.y = acc[1][nt][r] + bv;
      *(float2*)&out[rowo + ox] = v;
    }
  }
}

// ---------------------------------------------------------------------------
extern "C" void kernel_launch(void* const* d_in, const int* in_sizes, int n_in,
                              void* d_out, int out_size, void* d_ws, size_t ws_size,
                              hipStream_t stream) {
  const float* x    = (const float*)d_in[0];
  const float* W    = (const float*)d_in[1];
  const float* bias = (const float*)d_in[2];
  float* out = (float*)d_out;

  __bf16* xT = (__bf16*)d_ws;                            // 8,921,088 B
  __bf16* wa = (__bf16*)((char*)d_ws + WA_OFF);          // +524,288 B

  xpad_kernel  <<<NBATCH * XPAD, 256, 0, stream>>>(x, xT);   // 528 blocks
  upconv_wxform<<<64,           256, 0, stream>>>(W, wa);
  upconv_main  <<<1024,         256, 0, stream>>>(xT, wa, bias, out);
}
